// SelfAttention_39676907887534
// MI455X (gfx1250) — matrix operand order repaired and checked by reading, not compiled, not run on hardware
//
#include <hip/hip_runtime.h>

// ---------------- problem constants ----------------
#define BATCH   4
#define SEQ     2048
#define DIM     1024
#define HEADS   16
#define HDIM    64
#define BS      (BATCH * SEQ)         // 8192
#define SCALE   0.125f                // 64^-0.5

// ---------------- CDNA5 async global->LDS support ----------------
#if defined(__has_builtin)
#if __has_builtin(__builtin_amdgcn_global_load_async_to_lds_b128) && \
    __has_builtin(__builtin_amdgcn_s_wait_asynccnt)
#define HAVE_ASYNC_LDS 1
#endif
#endif

#define AS1 __attribute__((address_space(1)))
#define AS3 __attribute__((address_space(3)))

typedef int v4i_gcc __attribute__((vector_size(16)));   // builtin's pointee type

#if defined(HAVE_ASYNC_LDS)
#define ASYNC_CP_B128(gptr, lptr) \
    __builtin_amdgcn_global_load_async_to_lds_b128((AS1 v4i_gcc*)(gptr), (AS3 v4i_gcc*)(lptr), 0, 0)
#define WAIT_ASYNC(n) __builtin_amdgcn_s_wait_asynccnt(n)
#else
#define WAIT_ASYNC(n)
#endif

// ---------------- WMMA types ----------------
typedef __attribute__((ext_vector_type(16))) __bf16 bf16x16;
typedef __attribute__((ext_vector_type(8)))  float  v8f;

union Frag16 {
    bf16x16        v;
    uint4          q[2];
    unsigned int   u[8];
    unsigned short us[16];
};

__device__ __forceinline__ v8f wmma_bf16(const Frag16& a, const Frag16& b, v8f c) {
    return __builtin_amdgcn_wmma_f32_16x16x32_bf16(false, a.v, false, b.v,
                                                   (short)0, c, false, false);
}

__device__ __forceinline__ unsigned short f2bf(float f) {
    union { float f; unsigned int u; } v; v.f = f;
    unsigned int r = v.u + 0x7FFFu + ((v.u >> 16) & 1u);   // round-to-nearest-even
    return (unsigned short)(r >> 16);
}

// ---------------- fp32 -> bf16 conversion (SCALE folded into Q-rows of w_qkv) ----------------
__global__ void cvt_bf16_kernel(const float* __restrict__ src, unsigned short* __restrict__ dst,
                                long long n, float scale, long long scaleN) {
    long long i = (long long)blockIdx.x * blockDim.x + threadIdx.x;
    long long stride = (long long)gridDim.x * blockDim.x;
    for (; i < n; i += stride) {
        float v = src[i];
        if (i < scaleN) v *= scale;
        dst[i] = f2bf(v);
    }
}

// ---------------- tiled WMMA GEMM: C[M,N] = A[M,K] * B[N,K]^T ----------------
// mode 0: scatter bf16 into qkv [3][B][H][S][Hd];  mode 1: fp32 + bias -> out [M,N]
#define BM 128
#define BN 128
#define BK 64
#define LDT 72   // LDS row stride (ushorts): 64 + 8 pad (banks)

__global__ __launch_bounds__(256)
void gemm_bf16_wmma(const unsigned short* __restrict__ A,
                    const unsigned short* __restrict__ Bm,
                    int K, int N, int mode,
                    unsigned short* __restrict__ qkv_out,
                    float* __restrict__ f32_out,
                    const float* __restrict__ bias) {
#if defined(HAVE_ASYNC_LDS)
    __shared__ __align__(16) unsigned short As[2][BM * LDT];
    __shared__ __align__(16) unsigned short Bs[2][BN * LDT];
#else
    __shared__ __align__(16) unsigned short As[1][BM * LDT];
    __shared__ __align__(16) unsigned short Bs[1][BN * LDT];
#endif

    const int mBase = blockIdx.y * BM;
    const int nBase = blockIdx.x * BN;
    const int tid   = threadIdx.x;
    const int lane  = tid & 31;
    const int wid   = tid >> 5;             // 8 waves
    const int half  = lane >> 4;
    const int l16   = lane & 15;
    const int wm    = wid & 3;              // 4 waves along M (32 rows each)
    const int wn    = wid >> 2;             // 2 waves along N (64 cols each)

    v8f acc[2][4];
#pragma unroll
    for (int i = 0; i < 2; ++i)
#pragma unroll
        for (int j = 0; j < 4; ++j) acc[i][j] = (v8f)0.f;

    auto computeTile = [&](const unsigned short* as, const unsigned short* bs) {
#pragma unroll
        for (int ks = 0; ks < BK; ks += 32) {
            Frag16 af[2], bf[4];
#pragma unroll
            for (int mt = 0; mt < 2; ++mt) {
                const unsigned short* p = &as[(wm * 32 + mt * 16 + l16) * LDT + ks + half * 16];
                af[mt].q[0] = *(const uint4*)p;          // K = h*16 + 0..7
                af[mt].q[1] = *(const uint4*)(p + 8);    // K = h*16 + 8..15
            }
#pragma unroll
            for (int nt = 0; nt < 4; ++nt) {
                const unsigned short* p = &bs[(wn * 64 + nt * 16 + l16) * LDT + ks + half * 16];
                bf[nt].q[0] = *(const uint4*)p;          // K = h*16 + 0..7
                bf[nt].q[1] = *(const uint4*)(p + 8);    // K = h*16 + 8..15
            }
#pragma unroll
            for (int mt = 0; mt < 2; ++mt)
#pragma unroll
                for (int nt = 0; nt < 4; ++nt)
                    acc[mt][nt] = wmma_bf16(af[mt], bf[nt], acc[mt][nt]);
        }
    };

#if defined(HAVE_ASYNC_LDS)
    // ---------- double-buffered async DMA pipeline ----------
    auto issueTile = [&](int kb, int buf) {
#pragma unroll
        for (int i = 0; i < 4; ++i) {
            int c    = i * 256 + tid;              // 1024 chunks of 8 bf16 per matrix
            int row  = c >> 3;
            int col8 = (c & 7) * 8;
            ASYNC_CP_B128(A  + (size_t)(mBase + row) * K + kb + col8,
                          &As[buf][row * LDT + col8]);
            ASYNC_CP_B128(Bm + (size_t)(nBase + row) * K + kb + col8,
                          &Bs[buf][row * LDT + col8]);
        }
    };
    issueTile(0, 0);
    int buf = 0;
    for (int kb = 0; kb < K; kb += BK, buf ^= 1) {
        if (kb + BK < K) { issueTile(kb + BK, buf ^ 1); WAIT_ASYNC(8); }
        else             { WAIT_ASYNC(0); }
        __syncthreads();                 // everyone's tile `buf` resident
        computeTile(As[buf], Bs[buf]);
        __syncthreads();                 // done reading before it is re-filled
    }
#else
    // ---------- sync fallback: batch loads, then batch stores ----------
    for (int kb = 0; kb < K; kb += BK) {
        uint4 ra[4], rb[4];
#pragma unroll
        for (int i = 0; i < 4; ++i) {
            int c    = i * 256 + tid;
            int row  = c >> 3;
            int col8 = (c & 7) * 8;
            ra[i] = *(const uint4*)(A  + (size_t)(mBase + row) * K + kb + col8);
            rb[i] = *(const uint4*)(Bm + (size_t)(nBase + row) * K + kb + col8);
        }
#pragma unroll
        for (int i = 0; i < 4; ++i) {
            int c    = i * 256 + tid;
            int row  = c >> 3;
            int col8 = (c & 7) * 8;
            *(uint4*)&As[0][row * LDT + col8] = ra[i];
            *(uint4*)&Bs[0][row * LDT + col8] = rb[i];
        }
        __syncthreads();
        computeTile(As[0], Bs[0]);
        __syncthreads();
    }
#endif

    // writeback (C layout: row = r + 8*half, col = l16 within a 16x16 frag)
#pragma unroll
    for (int mt = 0; mt < 2; ++mt)
#pragma unroll
        for (int nt = 0; nt < 4; ++nt)
#pragma unroll
            for (int r = 0; r < 8; ++r) {
                int gm = mBase + wm * 32 + mt * 16 + r + 8 * half;
                int gn = nBase + wn * 64 + nt * 16 + l16;
                float v = acc[mt][nt][r];
                if (mode == 0) {
                    int which = gn >> 10, rem = gn & 1023;
                    int h = rem >> 6, hd = rem & 63;
                    int b = gm >> 11, s = gm & 2047;
                    size_t off = ((((size_t)which * BATCH + b) * HEADS + h) * SEQ + s) * HDIM + hd;
                    qkv_out[off] = f2bf(v);
                } else {
                    f32_out[(size_t)gm * N + gn] = v + bias[gn];
                }
            }
}

// ---------------- flash attention: one wave per 32-query tile ----------------
// qkv: bf16 [3][B][H][S][64] (Q pre-scaled by SCALE). obuf: bf16 [B][S][D] (head-merged)
__global__ __launch_bounds__(128)
void attn_wmma_kernel(const unsigned short* __restrict__ qkv,
                      unsigned short* __restrict__ obuf) {
    __shared__ __align__(16) unsigned short pS[4][32 * 40];      // P tile per wave (pad 40)
    __shared__ __align__(16) unsigned short vS[4][2][32 * 72];   // V tiles, double buffered

    const int lane = threadIdx.x & 31;
    const int wid  = threadIdx.x >> 5;
    const int half = lane >> 4;
    const int l16  = lane & 15;

    const int bh    = blockIdx.x;                 // b*16 + h
    const int qTile = blockIdx.y * 4 + wid;       // 0..63
    const int qBase = qTile * 32;

    const size_t hs = (size_t)SEQ * HDIM;         // 131072
    const unsigned short* Q  = qkv + (size_t)bh * hs;
    const unsigned short* Kp = qkv + (size_t)(BATCH * HEADS + bh) * hs;
    const unsigned short* Vp = qkv + (size_t)(2 * BATCH * HEADS + bh) * hs;

    unsigned short* psh = pS[wid];

    // stage one 32x64 V tile (one row per lane, 8 x b128)
    auto issueV = [&](int it, int vbuf) {
#if defined(HAVE_ASYNC_LDS)
#pragma unroll
        for (int i = 0; i < 8; ++i)
            ASYNC_CP_B128(Vp + (size_t)(it + lane) * HDIM + i * 8,
                          &vS[wid][vbuf][lane * 72 + i * 8]);
#else
        const uint4* src = (const uint4*)(Vp + (size_t)(it + lane) * HDIM);
#pragma unroll
        for (int i = 0; i < 8; ++i)
            *(uint4*)&vS[wid][vbuf][lane * 72 + i * 8] = src[i];
#endif
    };

    // resident Q A-fragments: 2 query sub-tiles x 2 k-steps over Hd
    Frag16 qa[2][2];
#pragma unroll
    for (int qt = 0; qt < 2; ++qt)
#pragma unroll
        for (int kt = 0; kt < 2; ++kt) {
            const unsigned short* p =
                Q + (size_t)(qBase + qt * 16 + l16) * HDIM + kt * 32 + half * 16;
            qa[qt][kt].q[0] = *(const uint4*)p;
            qa[qt][kt].q[1] = *(const uint4*)(p + 8);
        }

    float m[2][8], l[2][8];
    v8f accO[2][4];
#pragma unroll
    for (int qt = 0; qt < 2; ++qt) {
#pragma unroll
        for (int r = 0; r < 8; ++r) { m[qt][r] = -1e30f; l[qt][r] = 0.f; }
#pragma unroll
        for (int f = 0; f < 4; ++f) accO[qt][f] = (v8f)0.f;
    }

    issueV(0, 0);
    int vbuf = 0;
    for (int it = 0; it < SEQ; it += 32, vbuf ^= 1) {
        // K B-fragments straight from global; reused by both query sub-tiles
        Frag16 kb[2][2];
#pragma unroll
        for (int nt = 0; nt < 2; ++nt)
#pragma unroll
            for (int kt = 0; kt < 2; ++kt) {
                const unsigned short* p =
                    Kp + (size_t)(it + nt * 16 + l16) * HDIM + kt * 32 + half * 16;
                kb[nt][kt].q[0] = *(const uint4*)p;
                kb[nt][kt].q[1] = *(const uint4*)(p + 8);
            }

        // prefetch next V tile while we do QK^T + softmax
        if (it + 32 < SEQ) { issueV(it + 32, vbuf ^ 1); WAIT_ASYNC(8); }
        else               { WAIT_ASYNC(0); }

        // scores: S = Q * K^T (Q pre-scaled)
        v8f sc[2][2];
#pragma unroll
        for (int qt = 0; qt < 2; ++qt)
#pragma unroll
            for (int nt = 0; nt < 2; ++nt) {
                v8f z = (v8f)0.f;
                z = wmma_bf16(qa[qt][0], kb[nt][0], z);
                sc[qt][nt] = wmma_bf16(qa[qt][1], kb[nt][1], z);
            }

        // online softmax per query sub-tile (rows live across 16 lanes of each half)
#pragma unroll
        for (int qt = 0; qt < 2; ++qt) {
            float mp[8];
#pragma unroll
            for (int r = 0; r < 8; ++r) mp[r] = fmaxf(sc[qt][0][r], sc[qt][1][r]);
#pragma unroll
            for (int off = 1; off < 16; off <<= 1)
#pragma unroll
                for (int r = 0; r < 8; ++r)
                    mp[r] = fmaxf(mp[r], __shfl_xor(mp[r], off, 32));

            float ps[2][8], rs[8];
#pragma unroll
            for (int r = 0; r < 8; ++r) {
                float mn   = fmaxf(m[qt][r], mp[r]);
                float corr = __expf(m[qt][r] - mn);
                m[qt][r] = mn;
                ps[0][r] = __expf(sc[qt][0][r] - mn);
                ps[1][r] = __expf(sc[qt][1][r] - mn);
                rs[r] = ps[0][r] + ps[1][r];
                l[qt][r] *= corr;
#pragma unroll
                for (int f = 0; f < 4; ++f) accO[qt][f][r] *= corr;
            }
#pragma unroll
            for (int off = 1; off < 16; off <<= 1)
#pragma unroll
                for (int r = 0; r < 8; ++r)
                    rs[r] += __shfl_xor(rs[r], off, 32);
#pragma unroll
            for (int r = 0; r < 8; ++r) l[qt][r] += rs[r];

            // P (C-layout) -> LDS (rows qt*16 + ...)
#pragma unroll
            for (int nt = 0; nt < 2; ++nt)
#pragma unroll
                for (int r = 0; r < 8; ++r)
                    psh[(qt * 16 + r + 8 * half) * 40 + nt * 16 + l16] = f2bf(ps[nt][r]);
        }
        asm volatile("s_wait_dscnt 0" ::: "memory");

        // P A-fragments (one per query sub-tile)
        Frag16 pa[2];
#pragma unroll
        for (int qt = 0; qt < 2; ++qt) {
            const unsigned short* p = &psh[(qt * 16 + l16) * 40 + half * 16];
            pa[qt].q[0] = *(const uint4*)p;
            pa[qt].q[1] = *(const uint4*)(p + 8);
        }

        // V B-fragments gathered once, used by both query sub-tiles: O += P * V
        const unsigned short* vsh = vS[wid][vbuf];
#pragma unroll
        for (int f = 0; f < 4; ++f) {
            Frag16 vb;
            int hd = f * 16 + l16;
#pragma unroll
            for (int e = 0; e < 16; ++e)
                vb.us[e] = vsh[(half * 16 + e) * 72 + hd];
#pragma unroll
            for (int qt = 0; qt < 2; ++qt)
                accO[qt][f] = wmma_bf16(pa[qt], vb, accO[qt][f]);
        }
    }

    // epilogue: O / l -> bf16 [B][S][D]
    const int b = bh >> 4, h = bh & 15;
#pragma unroll
    for (int qt = 0; qt < 2; ++qt)
#pragma unroll
        for (int f = 0; f < 4; ++f)
#pragma unroll
            for (int r = 0; r < 8; ++r) {
                int q  = qBase + qt * 16 + r + 8 * half;
                int hd = f * 16 + l16;
                obuf[((size_t)(b * SEQ + q)) * DIM + h * HDIM + hd] =
                    f2bf(accO[qt][f][r] / l[qt][r]);
            }
}

// ---------------- host launcher ----------------
extern "C" void kernel_launch(void* const* d_in, const int* in_sizes, int n_in,
                              void* d_out, int out_size, void* d_ws, size_t ws_size,
                              hipStream_t stream) {
    (void)in_sizes; (void)n_in; (void)out_size; (void)ws_size;
    const float* x      = (const float*)d_in[0];
    const float* w_qkv  = (const float*)d_in[1];
    const float* w_proj = (const float*)d_in[2];
    const float* b_proj = (const float*)d_in[3];
    float* out = (float*)d_out;

    unsigned short* ws   = (unsigned short*)d_ws;
    unsigned short* xb   = ws;                               // [8192,1024]
    unsigned short* wqb  = xb  + (size_t)BS * DIM;           // [3072,1024]
    unsigned short* wpb  = wqb + (size_t)3 * DIM * DIM;      // [1024,1024]
    unsigned short* qkvb = wpb + (size_t)DIM * DIM;          // [3][B][H][S][64]
    unsigned short* ob   = qkvb + (size_t)3 * BS * DIM;      // [8192,1024]

    // fp32 -> bf16 (fold SCALE into the Q rows of w_qkv)
    cvt_bf16_kernel<<<2048, 256, 0, stream>>>(x, xb, (long long)BS * DIM, 1.f, 0LL);
    cvt_bf16_kernel<<<1024, 256, 0, stream>>>(w_qkv, wqb, (long long)3 * DIM * DIM,
                                              SCALE, (long long)DIM * DIM);
    cvt_bf16_kernel<<<512, 256, 0, stream>>>(w_proj, wpb, (long long)DIM * DIM, 1.f, 0LL);

    // QKV projection: [8192,3072] = xb * wqb^T, scattered to [3][B][H][S][Hd]
    dim3 g1(3 * DIM / BN, BS / BM);
    gemm_bf16_wmma<<<g1, 256, 0, stream>>>(xb, wqb, DIM, 3 * DIM, 0, qkvb, nullptr, nullptr);

    // attention: 64 (b,h) pairs x 16 blocks x 4 waves, 32 query rows per wave
    attn_wmma_kernel<<<dim3(BATCH * HEADS, SEQ / 32 / 4), 128, 0, stream>>>(qkvb, ob);

    // output projection + bias -> fp32
    dim3 g2(DIM / BN, BS / BM);
    gemm_bf16_wmma<<<g2, 256, 0, stream>>>(ob, wpb, DIM, DIM, 1, nullptr, out, b_proj);
}
